// VoxelBackBone8x_71184787964528
// MI455X (gfx1250) — matrix operand order repaired and checked by reading, not compiled, not run on hardware
//
#include <hip/hip_runtime.h>
#include <hip/hip_bf16.h>

// ---------------------------------------------------------------------------
// CDNA5 (gfx1250) WMMA implementation of the dense VoxelBackBone8x pipeline.
// Conv3D as implicit GEMM: D(16x16) = A(16x32) x B(32x16) + C, bf16 inputs,
// f32 accumulate, fused BN+ReLU+mask epilogue. Division-free hot kernel.
// Per-16-wide-W-strip occupancy tables let waves skip the GEMM entirely for
// strips whose output mask is all-zero (sparse voxel data: most strips).
// One wave computes all COUT/16 N-tiles, reusing the A fragment.
// ---------------------------------------------------------------------------

typedef __attribute__((ext_vector_type(16))) __bf16 v16bf;
typedef __attribute__((ext_vector_type(8)))  float  v8f;

union V16BF { v16bf v; __bf16 e[16]; float4 f4[2]; };

__device__ inline __bf16 f2bf(float f) { return (__bf16)f; }  // v_cvt, RTNE

constexpr int ilog2c(int x) { int r = 0; while (x > 1) { x >>= 1; ++r; } return r; }

// ---------------------------------------------------------------------------
// Utility kernels (cold path)
// ---------------------------------------------------------------------------
__global__ void fill_zero_kernel(float* __restrict__ p, long n) {
  long i = (long)blockIdx.x * blockDim.x + threadIdx.x;
  long stride = (long)gridDim.x * blockDim.x;
  for (; i < n; i += stride) p[i] = 0.0f;
}

__global__ void scatter_kernel(const float* __restrict__ feats,
                               const int* __restrict__ coords,
                               float* __restrict__ x, float* __restrict__ mask,
                               int N, int Dd, int Hh, int Ww, int Cpad, int Cin) {
  int i = blockIdx.x * blockDim.x + threadIdx.x;
  if (i >= N) return;
  int b = coords[4 * i + 0];
  int z = coords[4 * i + 1];
  int y = coords[4 * i + 2];
  int xx = coords[4 * i + 3];
  long pos = (((long)b * Dd + z) * Hh + y) * Ww + xx;
  for (int c = 0; c < Cin; ++c) x[pos * Cpad + c] = feats[(long)i * Cin + c];
  mask[pos] = 1.0f;
}

// occupancy conv -> binary mask at output resolution (grid mapped to geometry)
__global__ void mask_down_kernel(const float* __restrict__ mIn, float* __restrict__ mOut,
                                 int Din, int Hin, int Win,
                                 int Dout, int Hout, int Wout,
                                 int KD, int KH, int KW,
                                 int sd, int sh, int sw, int pd, int ph, int pw) {
  int ow = blockIdx.x * blockDim.x + threadIdx.x;
  if (ow >= Wout) return;
  int oh = blockIdx.y;
  int zz = blockIdx.z;                  // b*Dout + od, B==2
  int b = (zz >= Dout) ? 1 : 0;
  int od = zz - b * Dout;
  const float* mInB = mIn + (long)b * Din * Hin * Win;
  float any = 0.0f;
  for (int kd = 0; kd < KD; ++kd) {
    int id = od * sd - pd + kd;
    if (id < 0 || id >= Din) continue;
    for (int kh = 0; kh < KH; ++kh) {
      int ih = oh * sh - ph + kh;
      if (ih < 0 || ih >= Hin) continue;
      for (int kw = 0; kw < KW; ++kw) {
        int iw = ow * sw - pw + kw;
        if (iw < 0 || iw >= Win) continue;
        if (mInB[((long)id * Hin + ih) * Win + iw] > 0.0f) any = 1.0f;
      }
    }
  }
  mOut[(((long)b * Dout + od) * Hout + oh) * Wout + ow] = any;
}

// per-strip (16 W positions) occupancy: sOcc[row*tilesW + tw] = any(mask)
__global__ void strip_occ_kernel(const float* __restrict__ mask,
                                 float* __restrict__ sOcc,
                                 int Dout, int Hout, int Wout, int tilesW) {
  int tw = threadIdx.x;
  if (tw >= tilesW) return;
  int oh = blockIdx.y;
  int zz = blockIdx.z;
  int b = (zz >= Dout) ? 1 : 0;
  int od = zz - b * Dout;
  long row = ((long)b * Dout + od) * Hout + oh;
  long rowBase = row * Wout;
  int w0 = tw * 16;
  int w1 = w0 + 16 < Wout ? w0 + 16 : Wout;
  float any = 0.0f;
  for (int wq = w0; wq < w1; ++wq)
    if (mask[rowBase + wq] > 0.0f) any = 1.0f;
  sOcc[row * tilesW + tw] = any;
}

// repack weights (co,ci,kd,kh,kw) f32 -> [tap][co][ci_pad] bf16, zero pad ci
// channel counts are powers of two -> shift/mask decode
__global__ void pack_w_kernel(const float* __restrict__ w, __bf16* __restrict__ pk,
                              int CiAct, int K, int lgCpad, int lgCout) {
  long i = (long)blockIdx.x * blockDim.x + threadIdx.x;
  long n = (long)K << (lgCpad + lgCout);
  if (i >= n) return;
  int ci = (int)(i & ((1 << lgCpad) - 1));
  int co = (int)((i >> lgCpad) & ((1 << lgCout) - 1));
  int tap = (int)(i >> (lgCpad + lgCout));
  float v = (ci < CiAct) ? w[((long)co * CiAct + ci) * K + tap] : 0.0f;
  pk[i] = f2bf(v);
}

// ---------------------------------------------------------------------------
// WMMA implicit-GEMM conv kernel (templated, division-free)
//   activations: NDHW x CPAD (f32), weights packed [tap][COUT][CPAD] bf16
//   grid = (tilesW, Hout, 2*Dout), block = 32: one wave handles a 16-wide W
//   strip and ALL COUT/16 N-tiles (A fragment loaded once per K-segment).
// ---------------------------------------------------------------------------
template <int KHc, int KWc, int CPAD>
__device__ inline void loadA_seg(const float* __restrict__ inB, bool rowValid,
                                 int id0, int ih0, int iw0,
                                 int tap, bool segValid,
                                 int Din, int Hin, int Win,
                                 int hv, __bf16* dst) {
  float v[8] = {0.f, 0.f, 0.f, 0.f, 0.f, 0.f, 0.f, 0.f};
  if (segValid && rowValid) {
    constexpr int KHW = KHc * KWc;
    int kd = tap / KHW;             // constant division -> mul/shift
    int r = tap - kd * KHW;
    int kh = r / KWc;
    int kw = r - kh * KWc;
    int id = id0 + kd, ih = ih0 + kh, iw = iw0 + kw;
    if (id >= 0 && id < Din && ih >= 0 && ih < Hin && iw >= 0 && iw < Win) {
      const float* p = inB + (((long)id * Hin + ih) * (long)Win + iw) * CPAD
                           + hv * 8;
      float4 x0 = ((const float4*)p)[0];
      float4 x1 = ((const float4*)p)[1];
      v[0] = x0.x; v[1] = x0.y; v[2] = x0.z; v[3] = x0.w;
      v[4] = x1.x; v[5] = x1.y; v[6] = x1.z; v[7] = x1.w;
    }
  }
#pragma unroll
  for (int j = 0; j < 8; ++j) dst[j] = f2bf(v[j]);
}

template <int CPAD, int COUT, int KD, int KH, int KW, int OUT_NCDHW>
__global__ void conv_wmma_t(
    const float* __restrict__ in, float* __restrict__ out,
    const __bf16* __restrict__ wpk,
    const float* __restrict__ gamma, const float* __restrict__ beta,
    const float* __restrict__ mean, const float* __restrict__ var,
    const float* __restrict__ outMask, const float* __restrict__ stripOcc,
    int tilesW,
    int Din, int Hin, int Win, int Dout, int Hout, int Wout,
    int sd, int sh, int sw, int pd, int ph, int pw) {
  constexpr int TILES_N = COUT / 16;
  constexpr int SEGS_PER_TAP = CPAD / 16;
  constexpr int TOTAL_SEGS = KD * KH * KW * SEGS_PER_TAP;

  const int lane = threadIdx.x;               // blockDim == 32
  const int m = lane & 15;
  const int hv = lane >> 4;
  const int n = m;

  const int tileW = blockIdx.x;
  const int oh = blockIdx.y;
  const int zz = blockIdx.z;                  // b*Dout + od, B==2
  const int b = (zz >= Dout) ? 1 : 0;
  const int od = zz - b * Dout;

  const long row = ((long)b * Dout + od) * Hout + oh;
  const long rowBase = row * Wout;

  // ---- strip-level sparsity skip (wave-uniform) ----
  const bool live = stripOcc[row * tilesW + tileW] > 0.0f;
  if (!live) {
    // masked output is identically zero for this strip: store zeros
#pragma unroll
    for (int tN = 0; tN < TILES_N; ++tN) {
      const int ng = tN * 16 + n;
#pragma unroll
      for (int v = 0; v < 8; ++v) {
        const int ow_r = tileW * 16 + hv * 8 + v;
        if (ow_r < Wout) {
          if (OUT_NCDHW)
            out[((((long)b * COUT + ng) * Dout + od) * Hout + oh) * Wout + ow_r] = 0.0f;
          else
            out[(rowBase + ow_r) * COUT + ng] = 0.0f;
        }
      }
    }
    return;
  }

  const int ow = tileW * 16 + m;              // A-row coordinate for this lane
  const bool rowValid = ow < Wout;

  const int id0 = od * sd - pd;
  const int ih0 = oh * sh - ph;
  const int iw0 = ow * sw - pw;
  const float* inB = in + (long)b * Din * Hin * Win * CPAD;

  v8f acc[TILES_N];
#pragma unroll
  for (int tN = 0; tN < TILES_N; ++tN)
    acc[tN] = (v8f){0.f, 0.f, 0.f, 0.f, 0.f, 0.f, 0.f, 0.f};

  for (int s = 0; s < TOTAL_SEGS; s += 2) {
    V16BF Aa;
    int tapB0, cibB;                   // B addressing for this lane's K-half
    if constexpr (SEGS_PER_TAP == 1) {
      // segments are whole taps; 27 taps -> one zero A-segment at the end.
      // (zero A half guarantees zero contribution, so B may load tap 0.)
      const int tap0 = s;
      const bool v1 = (s + 1) < TOTAL_SEGS;
      const int tap1 = v1 ? s + 1 : 0;
      loadA_seg<KH, KW, CPAD>(inB, rowValid, id0, ih0, iw0, hv ? tap1 : tap0, hv ? v1 : true,
                              Din, Hin, Win, 0, &Aa.e[0]);
      loadA_seg<KH, KW, CPAD>(inB, rowValid, id0, ih0, iw0, hv ? tap1 : tap0, hv ? v1 : true,
                              Din, Hin, Win, 1, &Aa.e[8]);
      tapB0 = hv ? tap1 : tap0;
      cibB = 0;
    } else {
      // both 16-wide segments live in the same tap: one address/bounds check
      constexpr int KHW = KH * KW;
      const int tap = s / SEGS_PER_TAP;       // constant division
      const int cib = (s & (SEGS_PER_TAP - 1)) * 16;
      const int kd = tap / KHW;
      const int r = tap - kd * KHW;
      const int kh = r / KW;
      const int kw = r - kh * KW;
      const int id = id0 + kd, ih = ih0 + kh, iw = iw0 + kw;
      float v[16];
#pragma unroll
      for (int j = 0; j < 16; ++j) v[j] = 0.0f;
      if (rowValid && id >= 0 && id < Din && ih >= 0 && ih < Hin &&
          iw >= 0 && iw < Win) {
        const float* p = inB + (((long)id * Hin + ih) * (long)Win + iw) * CPAD
                             + cib + hv * 16;
        float4 x0 = ((const float4*)p)[0];
        float4 x1 = ((const float4*)p)[1];
        float4 x2 = ((const float4*)(p + 8))[0];
        float4 x3 = ((const float4*)(p + 8))[1];
        v[0] = x0.x; v[1] = x0.y; v[2]  = x0.z; v[3]  = x0.w;
        v[4] = x1.x; v[5] = x1.y; v[6]  = x1.z; v[7]  = x1.w;
        v[8] = x2.x; v[9] = x2.y; v[10] = x2.z; v[11] = x2.w;
        v[12] = x3.x; v[13] = x3.y; v[14] = x3.z; v[15] = x3.w;
      }
#pragma unroll
      for (int j = 0; j < 16; ++j) Aa.e[j] = f2bf(v[j]);
      tapB0 = tap;
      cibB = cib + hv * 16;
    }

    // all N-tiles reuse the A fragment
#pragma unroll
    for (int tN = 0; tN < TILES_N; ++tN) {
      V16BF Bb;
      const __bf16* p = wpk + ((long)tapB0 * COUT + tN * 16 + n) * CPAD + cibB;
      Bb.f4[0] = ((const float4*)p)[0];
      Bb.f4[1] = ((const float4*)p)[1];
      acc[tN] = __builtin_amdgcn_wmma_f32_16x16x32_bf16(
          /*neg_a=*/false, Aa.v, /*neg_b=*/false, Bb.v,
          /*c_mod=*/(short)0, acc[tN], /*reuse_a=*/false, /*reuse_b=*/false);
    }
  }

  // fused BN + ReLU + mask epilogue; C/D layout: row = hv*8 + v, col = n
#pragma unroll
  for (int tN = 0; tN < TILES_N; ++tN) {
    const int ng = tN * 16 + n;
    const float sc = gamma[ng] * rsqrtf(var[ng] + 1e-3f);
    const float bi = beta[ng] - mean[ng] * sc;
#pragma unroll
    for (int v = 0; v < 8; ++v) {
      const int ow_r = tileW * 16 + hv * 8 + v;
      if (ow_r < Wout) {
        float y = acc[tN][v] * sc + bi;
        y = y > 0.0f ? y : 0.0f;
        y *= outMask[rowBase + ow_r];
        if (OUT_NCDHW)
          out[((((long)b * COUT + ng) * Dout + od) * Hout + oh) * Wout + ow_r] = y;
        else
          out[(rowBase + ow_r) * COUT + ng] = y;
      }
    }
  }
}

// ---------------------------------------------------------------------------
// Host-side orchestration
// ---------------------------------------------------------------------------
template <int CPAD, int COUT, int KD, int KH, int KW, int OUT_NCDHW>
static void run_conv(hipStream_t s, const float* in, float* out,
                     const float* w, const float* g, const float* be,
                     const float* mn, const float* vr,
                     const float* outMask, const float* stripOcc,
                     __bf16* wpk, int CiAct,
                     int Din, int Hin, int Win,
                     int Dout, int Hout, int Wout,
                     int sd, int sh, int sw, int pd, int ph, int pw) {
  const int K = KD * KH * KW;
  const long pkN = (long)K * COUT * CPAD;
  pack_w_kernel<<<(unsigned)((pkN + 255) / 256), 256, 0, s>>>(
      w, wpk, CiAct, K, ilog2c(CPAD), ilog2c(COUT));
  const int tilesW = (Wout + 15) / 16;
  dim3 grid((unsigned)tilesW, (unsigned)Hout, (unsigned)(2 * Dout));
  conv_wmma_t<CPAD, COUT, KD, KH, KW, OUT_NCDHW><<<grid, 32, 0, s>>>(
      in, out, wpk, g, be, mn, vr, outMask, stripOcc, tilesW,
      Din, Hin, Win, Dout, Hout, Wout, sd, sh, sw, pd, ph, pw);
}

static void run_mask(hipStream_t s, const float* mIn, float* mOut, float* sOcc,
                     int Din, int Hin, int Win, int Dout, int Hout, int Wout,
                     int KD, int KH, int KW,
                     int sd, int sh, int sw, int pd, int ph, int pw) {
  dim3 grid((unsigned)((Wout + 63) / 64), (unsigned)Hout, (unsigned)(2 * Dout));
  mask_down_kernel<<<grid, 64, 0, s>>>(mIn, mOut, Din, Hin, Win, Dout, Hout, Wout,
                                       KD, KH, KW, sd, sh, sw, pd, ph, pw);
  const int tilesW = (Wout + 15) / 16;
  dim3 sg(1, (unsigned)Hout, (unsigned)(2 * Dout));
  strip_occ_kernel<<<sg, 16, 0, s>>>(mOut, sOcc, Dout, Hout, Wout, tilesW);
}

extern "C" void kernel_launch(void* const* d_in, const int* in_sizes, int n_in,
                              void* d_out, int out_size, void* d_ws, size_t ws_size,
                              hipStream_t stream) {
  (void)in_sizes; (void)n_in; (void)out_size; (void)ws_size;
  const float* feats  = (const float*)d_in[0];
  const int*   coords = (const int*)d_in[1];
  auto P = [&](int layer, int j) -> const float* {
    return (const float*)d_in[2 + layer * 5 + j];  // j: 0=w 1=gamma 2=beta 3=mean 4=var
  };
  enum { L_conv_in = 0, L_subm1, L_sp2, L_subm2a, L_subm2b, L_sp3, L_subm3a,
         L_subm3b, L_sp4, L_subm4a, L_subm4b, L_out };

  const int Bn = 2, N = 150000;
  const int D0 = 41, H0 = 200, W0 = 200;
  const int D1 = 21, H1 = 100, W1 = 100;
  const int D2 = 11, H2 = 50,  W2 = 50;
  const int D3 = 5,  H3 = 25,  W3 = 25;
  const int D4 = 2,  H4 = 25,  W4 = 25;
  const long NP0 = (long)Bn * D0 * H0 * W0;  // 3,280,000
  const long NP1 = (long)Bn * D1 * H1 * W1;  //   420,000
  const long NP2 = (long)Bn * D2 * H2 * W2;  //    55,000
  const long NP3 = (long)Bn * D3 * H3 * W3;  //     6,250
  const long NP4 = (long)Bn * D4 * H4 * W4;  //     2,500
  const int TW0 = 13, TW1 = 7, TW2 = 4, TW3 = 2, TW4 = 2;   // ceil(W/16)
  const long NS0 = (long)Bn * D0 * H0 * TW0;
  const long NS1 = (long)Bn * D1 * H1 * TW1;
  const long NS2 = (long)Bn * D2 * H2 * TW2;
  const long NS3 = (long)Bn * D3 * H3 * TW3;
  const long NS4 = (long)Bn * D4 * H4 * TW4;

  float* wsf  = (float*)d_ws;
  float* bufA = wsf;
  float* bufB = bufA + NP0 * 16;
  float* m0   = bufB + NP0 * 16;
  float* m1   = m0 + NP0;
  float* m2   = m1 + NP1;
  float* m3   = m2 + NP2;
  float* m4   = m3 + NP3;
  float* s0   = m4 + NP4;
  float* s1   = s0 + NS0;
  float* s2   = s1 + NS1;
  float* s3   = s2 + NS2;
  float* s4   = s3 + NS3;
  __bf16* wpk = (__bf16*)(s4 + NS4);

  // scatter voxels into padded 16-channel dense grid + occupancy mask
  fill_zero_kernel<<<4096, 256, 0, stream>>>(bufA, NP0 * 16);
  fill_zero_kernel<<<1024, 256, 0, stream>>>(m0, NP0);
  scatter_kernel<<<(N + 255) / 256, 256, 0, stream>>>(feats, coords, bufA, m0,
                                                      N, D0, H0, W0, 16, 4);
  { dim3 sg(1, (unsigned)H0, (unsigned)(2 * D0));
    strip_occ_kernel<<<sg, 16, 0, stream>>>(m0, s0, D0, H0, W0, TW0); }

  // conv_in (4->16) subm @ scale0, A->B
  run_conv<16, 16, 3, 3, 3, 0>(stream, bufA, bufB, P(L_conv_in,0), P(L_conv_in,1),
      P(L_conv_in,2), P(L_conv_in,3), P(L_conv_in,4), m0, s0, wpk, 4,
      D0,H0,W0, D0,H0,W0, 1,1,1, 1,1,1);
  // subm1 (16->16) B->A
  run_conv<16, 16, 3, 3, 3, 0>(stream, bufB, bufA, P(L_subm1,0), P(L_subm1,1),
      P(L_subm1,2), P(L_subm1,3), P(L_subm1,4), m0, s0, wpk, 16,
      D0,H0,W0, D0,H0,W0, 1,1,1, 1,1,1);

  // sp2 (16->32) stride2 pad1, A->B; new mask m1
  run_mask(stream, m0, m1, s1, D0,H0,W0, D1,H1,W1, 3,3,3, 2,2,2, 1,1,1);
  run_conv<16, 32, 3, 3, 3, 0>(stream, bufA, bufB, P(L_sp2,0), P(L_sp2,1),
      P(L_sp2,2), P(L_sp2,3), P(L_sp2,4), m1, s1, wpk, 16,
      D0,H0,W0, D1,H1,W1, 2,2,2, 1,1,1);
  // subm2a (32->32) B->A ; subm2b A->B
  run_conv<32, 32, 3, 3, 3, 0>(stream, bufB, bufA, P(L_subm2a,0), P(L_subm2a,1),
      P(L_subm2a,2), P(L_subm2a,3), P(L_subm2a,4), m1, s1, wpk, 32,
      D1,H1,W1, D1,H1,W1, 1,1,1, 1,1,1);
  run_conv<32, 32, 3, 3, 3, 0>(stream, bufA, bufB, P(L_subm2b,0), P(L_subm2b,1),
      P(L_subm2b,2), P(L_subm2b,3), P(L_subm2b,4), m1, s1, wpk, 32,
      D1,H1,W1, D1,H1,W1, 1,1,1, 1,1,1);

  // sp3 (32->64) stride2 pad1, B->A; new mask m2
  run_mask(stream, m1, m2, s2, D1,H1,W1, D2,H2,W2, 3,3,3, 2,2,2, 1,1,1);
  run_conv<32, 64, 3, 3, 3, 0>(stream, bufB, bufA, P(L_sp3,0), P(L_sp3,1),
      P(L_sp3,2), P(L_sp3,3), P(L_sp3,4), m2, s2, wpk, 32,
      D1,H1,W1, D2,H2,W2, 2,2,2, 1,1,1);
  // subm3a (64->64) A->B ; subm3b B->A
  run_conv<64, 64, 3, 3, 3, 0>(stream, bufA, bufB, P(L_subm3a,0), P(L_subm3a,1),
      P(L_subm3a,2), P(L_subm3a,3), P(L_subm3a,4), m2, s2, wpk, 64,
      D2,H2,W2, D2,H2,W2, 1,1,1, 1,1,1);
  run_conv<64, 64, 3, 3, 3, 0>(stream, bufB, bufA, P(L_subm3b,0), P(L_subm3b,1),
      P(L_subm3b,2), P(L_subm3b,3), P(L_subm3b,4), m2, s2, wpk, 64,
      D2,H2,W2, D2,H2,W2, 1,1,1, 1,1,1);

  // sp4 (64->64) stride2 pad(0,1,1), A->B; new mask m3
  run_mask(stream, m2, m3, s3, D2,H2,W2, D3,H3,W3, 3,3,3, 2,2,2, 0,1,1);
  run_conv<64, 64, 3, 3, 3, 0>(stream, bufA, bufB, P(L_sp4,0), P(L_sp4,1),
      P(L_sp4,2), P(L_sp4,3), P(L_sp4,4), m3, s3, wpk, 64,
      D2,H2,W2, D3,H3,W3, 2,2,2, 0,1,1);
  // subm4a (64->64) B->A ; subm4b A->B
  run_conv<64, 64, 3, 3, 3, 0>(stream, bufB, bufA, P(L_subm4a,0), P(L_subm4a,1),
      P(L_subm4a,2), P(L_subm4a,3), P(L_subm4a,4), m3, s3, wpk, 64,
      D3,H3,W3, D3,H3,W3, 1,1,1, 1,1,1);
  run_conv<64, 64, 3, 3, 3, 0>(stream, bufA, bufB, P(L_subm4b,0), P(L_subm4b,1),
      P(L_subm4b,2), P(L_subm4b,3), P(L_subm4b,4), m3, s3, wpk, 64,
      D3,H3,W3, D3,H3,W3, 1,1,1, 1,1,1);

  // out (64->128) k(3,1,1) s(2,1,1) p0, B -> d_out (NCDHW); new mask m4
  run_mask(stream, m3, m4, s4, D3,H3,W3, D4,H4,W4, 3,1,1, 2,1,1, 0,0,0);
  run_conv<64, 128, 3, 1, 1, 1>(stream, bufB, (float*)d_out, P(L_out,0), P(L_out,1),
      P(L_out,2), P(L_out,3), P(L_out,4), m4, s4, wpk, 64,
      D3,H3,W3, D4,H4,W4, 2,1,1, 0,0,0);
}
